// ContinuousTimeHopfieldLayer_88510686036320
// MI455X (gfx1250) — compile-verified, matrix operand order checked by reading, not run on hardware
//
#include <hip/hip_runtime.h>
#include <hip/hip_bf16.h>

// ---------------------------------------------------------------------------
// Continuous-time Hopfield layer, RKF45 integrator, MI455X (gfx1250).
// GEMMs run on v_wmma_f32_16x16x32_bf16 (f32 accumulate). Working set
// (~130MB) fits in 192MB L2; problem is matrix-core bound, so bf16 WMMA
// with fp32 accumulation is the right precision/throughput point.
// Wave tile 32x64 -> 8 WMMAs per 6 LDS fragment loads (1.33x the math per
// LDS byte vs a 32x32 tile). Early-stop is a device-side flag; all
// kernels early-exit uniformly (EXEC all-ones when WMMA executes).
// ---------------------------------------------------------------------------

typedef __attribute__((ext_vector_type(16))) __bf16 v16bf;
typedef __attribute__((ext_vector_type(8)))  float  v8f;

#define HOP_M   4096
#define HOP_N   1024   // state width == K == N of the weight matrix
#define HOP_DT  0.1f
#define HOP_TOL 1e-3f
#define HOP_STEPS 100

#define MT 128         // M tile per workgroup
#define NT 128         // N tile per workgroup
#define KT 32          // K slice == WMMA K
#define ASTR 34        // LDS row stride in bf16 elems (32 + 2 pad -> 17-bank stride)
#define BSTR 34

__device__ __forceinline__ unsigned short hop_f2bf(float f) {
    // round-to-nearest-even fp32 -> bf16
    unsigned u = __float_as_uint(f);
    unsigned r = u + 0x7FFFu + ((u >> 16) & 1u);
    return (unsigned short)(r >> 16);
}

union HopFrag { unsigned u[8]; v16bf v; };

// ---------------------------------------------------------------------------
// init: z = x, header (done, amax) = 0
// ---------------------------------------------------------------------------
__global__ void hop_init(const float* __restrict__ x, float* __restrict__ z,
                         int n, int* __restrict__ hdr) {
    int i = blockIdx.x * blockDim.x + threadIdx.x;
    if (i == 0) { hdr[0] = 0; ((unsigned*)hdr)[1] = 0u; }
    int stride = gridDim.x * blockDim.x;
    for (; i < n; i += stride) z[i] = x[i];
}

// ---------------------------------------------------------------------------
// prep: Wsb = bf16( sym_zero_diag(W) ), row-major (symmetric, so also
// column-major — lets the GEMM read B tiles n-major with coalesced loads)
// ---------------------------------------------------------------------------
__global__ void hop_prep(const float* __restrict__ W,
                         unsigned short* __restrict__ Wsb) {
    int i = blockIdx.x * blockDim.x + threadIdx.x;
    int stride = gridDim.x * blockDim.x;
    int total = HOP_N * HOP_N;
    for (; i < total; i += stride) {
        int r = i >> 10, c = i & (HOP_N - 1);
        float v = (r == c) ? 0.f
                           : 0.5f * (W[(size_t)r * HOP_N + c] + W[(size_t)c * HOP_N + r]);
        Wsb[i] = hop_f2bf(v);
    }
}

// ---------------------------------------------------------------------------
// Fused GEMM + pvf epilogue:  Kout = tanh(Ain @ Ws + b) - Ain
// Optionally atomically max-reduces |Kout| into amax (positive-float bit trick).
// 8 waves, wave grid 4(M) x 2(N), wave tile 32x64, 8 WMMA per K slice.
// ---------------------------------------------------------------------------
__global__ __launch_bounds__(256)
void hop_gemm_pvf(const float* __restrict__ Ain,
                  const unsigned short* __restrict__ Wsb,
                  const float* __restrict__ bvec,
                  float* __restrict__ Kout,
                  unsigned* __restrict__ amax,
                  int do_amax,
                  const int* __restrict__ done_flag) {
    if (*(volatile const int*)done_flag) return;

    __shared__ unsigned short As[MT * ASTR];
    __shared__ unsigned short Bs[NT * BSTR];

    const int tid  = threadIdx.x;
    const int lane = tid & 31;
    const int wave = tid >> 5;
    const int wm   = wave >> 1;            // 0..3  (M position of wave, 32 rows)
    const int wn   = wave & 1;             // 0..1  (N position of wave, 64 cols)
    const int m0   = blockIdx.y * MT;
    const int n0   = blockIdx.x * NT;

    const int half = lane >> 4;
    const int l16  = lane & 15;

    v8f acc[2][4];
#pragma unroll
    for (int i = 0; i < 2; ++i)
#pragma unroll
        for (int j = 0; j < 4; ++j) acc[i][j] = (v8f){};

    // cooperative tile-load coordinates
    const int arow = tid >> 1;             // 0..127
    const int acol = (tid & 1) * 16;       // 16 floats per thread
    const int brow = tid >> 1;             // 0..127 (n within tile)
    const int bcol = (tid & 1) * 16;       // 16 bf16 (32B) per thread

    for (int k0 = 0; k0 < HOP_N; k0 += KT) {
        // ---- A tile: fp32 state -> bf16 LDS ----
        const float* ag = Ain + (size_t)(m0 + arow) * HOP_N + k0 + acol;
        if (k0 + KT < HOP_N) __builtin_prefetch(ag + KT, 0, 1);  // global_prefetch_b8
        unsigned short* as = &As[arow * ASTR + acol];
#pragma unroll
        for (int i = 0; i < 4; ++i) {
            float4 f = ((const float4*)ag)[i];
            *(unsigned*)(as + i * 4 + 0) =
                (unsigned)hop_f2bf(f.x) | ((unsigned)hop_f2bf(f.y) << 16);
            *(unsigned*)(as + i * 4 + 2) =
                (unsigned)hop_f2bf(f.z) | ((unsigned)hop_f2bf(f.w) << 16);
        }
        // ---- B tile: symmetric Ws, read row n (== column n), k-contiguous ----
        const unsigned short* bg = Wsb + (size_t)(n0 + brow) * HOP_N + k0 + bcol;
        unsigned short* bs = &Bs[brow * BSTR + bcol];
#pragma unroll
        for (int i = 0; i < 2; ++i) {
            const uint4 q = ((const uint4*)bg)[i];
            *(unsigned*)(bs + i * 8 + 0) = q.x;
            *(unsigned*)(bs + i * 8 + 2) = q.y;
            *(unsigned*)(bs + i * 8 + 4) = q.z;
            *(unsigned*)(bs + i * 8 + 6) = q.w;
        }

        __syncthreads();

        // ---- gather WMMA fragments (ISA 16-bit A layout, mirrored for B) ----
        HopFrag a[2], b[4];
#pragma unroll
        for (int j = 0; j < 8; ++j) {
            const int kk = ((j & 3) << 1) + ((j >> 2) << 4) + (half << 3);
            a[0].u[j] = *(const unsigned*)&As[(wm * 32 + 0  + l16) * ASTR + kk];
            a[1].u[j] = *(const unsigned*)&As[(wm * 32 + 16 + l16) * ASTR + kk];
            b[0].u[j] = *(const unsigned*)&Bs[(wn * 64 + 0  + l16) * BSTR + kk];
            b[1].u[j] = *(const unsigned*)&Bs[(wn * 64 + 16 + l16) * BSTR + kk];
            b[2].u[j] = *(const unsigned*)&Bs[(wn * 64 + 32 + l16) * BSTR + kk];
            b[3].u[j] = *(const unsigned*)&Bs[(wn * 64 + 48 + l16) * BSTR + kk];
        }
#pragma unroll
        for (int i = 0; i < 2; ++i)
#pragma unroll
            for (int j = 0; j < 4; ++j)
                acc[i][j] = __builtin_amdgcn_wmma_f32_16x16x32_bf16(
                    false, a[i].v, false, b[j].v, (short)0, acc[i][j], false, false);

        __syncthreads();
    }

    // ---- epilogue: k = tanh(acc + b[n]) - Ain[m][n], optional amax ----
    const int mbase = m0 + wm * 32 + half * 8;
    const int nbase = n0 + wn * 64 + l16;
    float local = 0.f;
#pragma unroll
    for (int j = 0; j < 4; ++j) {
        const int n  = nbase + j * 16;
        const float bn = bvec[n];
#pragma unroll
        for (int i = 0; i < 2; ++i) {
#pragma unroll
            for (int r = 0; r < 8; ++r) {
                const int m = mbase + i * 16 + r;
                const size_t idx = (size_t)m * HOP_N + n;
                float o = tanhf(acc[i][j][r] + bn) - Ain[idx];
                Kout[idx] = o;
                local = fmaxf(local, fabsf(o));
            }
        }
    }
    if (do_amax) atomicMax(amax, __float_as_uint(local));
}

// ---------------------------------------------------------------------------
// flag: done |= (amax < tol); amax = 0 for the next step
// ---------------------------------------------------------------------------
__global__ void hop_flag(int* hdr) {
    if (threadIdx.x == 0) {
        float m = __uint_as_float(((volatile unsigned*)hdr)[1]);
        if (m < HOP_TOL) hdr[0] = 1;
        ((unsigned*)hdr)[1] = 0u;
    }
}

// ---------------------------------------------------------------------------
// combine: dst = base + DT * (c0*p0 + ... + c4*p4)   (dst may alias base)
// ---------------------------------------------------------------------------
__global__ void hop_combine(float* __restrict__ dst, const float* __restrict__ base,
                            const float* __restrict__ p0, const float* __restrict__ p1,
                            const float* __restrict__ p2, const float* __restrict__ p3,
                            const float* __restrict__ p4,
                            float c0, float c1, float c2, float c3, float c4,
                            int n, const int* __restrict__ done_flag) {
    if (*(volatile const int*)done_flag) return;
    int i = blockIdx.x * blockDim.x + threadIdx.x;
    int stride = gridDim.x * blockDim.x;
    for (; i < n; i += stride) {
        float s = c0 * p0[i];
        if (c1 != 0.f) s += c1 * p1[i];
        if (c2 != 0.f) s += c2 * p2[i];
        if (c3 != 0.f) s += c3 * p3[i];
        if (c4 != 0.f) s += c4 * p4[i];
        dst[i] = base[i] + HOP_DT * s;
    }
}

// ---------------------------------------------------------------------------
extern "C" void kernel_launch(void* const* d_in, const int* in_sizes, int n_in,
                              void* d_out, int out_size, void* d_ws, size_t ws_size,
                              hipStream_t stream) {
    const float* x = (const float*)d_in[0];
    const float* W = (const float*)d_in[1];
    const float* b = (const float*)d_in[2];
    float* z = (float*)d_out;                       // state lives in d_out
    const int elems = HOP_M * HOP_N;                // 4,194,304

    char* ws = (char*)d_ws;
    int* hdr = (int*)ws;                            // [0]=done, [1]=amax bits
    unsigned short* Wsb = (unsigned short*)(ws + 256);
    float* y = (float*)(ws + 256 + (size_t)HOP_N * HOP_N * sizeof(unsigned short));
    float* kb[6];
    for (int s = 0; s < 6; ++s) kb[s] = y + (size_t)elems * (1 + s);

    hop_init<<<4096, 256, 0, stream>>>(x, z, elems, hdr);
    hop_prep<<<1024, 256, 0, stream>>>(W, Wsb);

    const dim3 gemm_grid(HOP_N / NT, HOP_M / MT);   // 8 x 32 workgroups
    const int CB = 4096, CT = 256;                  // combine launch config
    unsigned* amax = (unsigned*)(hdr + 1);

    // Fehlberg tableau
    const float A21 = 0.25f;
    const float A31 = 3.f / 32.f,      A32 = 9.f / 32.f;
    const float A41 = 1932.f / 2197.f, A42 = -7200.f / 2197.f, A43 = 7296.f / 2197.f;
    const float A51 = 439.f / 216.f,   A52 = -8.f,  A53 = 3680.f / 513.f, A54 = -845.f / 4104.f;
    const float A61 = -8.f / 27.f,     A62 = 2.f,   A63 = -3544.f / 2565.f,
                A64 = 1859.f / 4104.f, A65 = -11.f / 40.f;
    const float B1 = 16.f / 135.f, B3 = 6656.f / 12825.f, B4 = 28561.f / 56430.f,
                B5 = -9.f / 50.f,  B6 = 2.f / 55.f;

    for (int step = 0; step < HOP_STEPS; ++step) {
        // k1 = pvf(z), with global max|k1| reduction; then evaluate stop flag
        hop_gemm_pvf<<<gemm_grid, 256, 0, stream>>>(z, Wsb, b, kb[0], amax, 1, hdr);
        hop_flag<<<1, 32, 0, stream>>>(hdr);

        hop_combine<<<CB, CT, 0, stream>>>(y, z, kb[0], kb[0], kb[0], kb[0], kb[0],
                                           A21, 0.f, 0.f, 0.f, 0.f, elems, hdr);
        hop_gemm_pvf<<<gemm_grid, 256, 0, stream>>>(y, Wsb, b, kb[1], amax, 0, hdr);

        hop_combine<<<CB, CT, 0, stream>>>(y, z, kb[0], kb[1], kb[0], kb[0], kb[0],
                                           A31, A32, 0.f, 0.f, 0.f, elems, hdr);
        hop_gemm_pvf<<<gemm_grid, 256, 0, stream>>>(y, Wsb, b, kb[2], amax, 0, hdr);

        hop_combine<<<CB, CT, 0, stream>>>(y, z, kb[0], kb[1], kb[2], kb[0], kb[0],
                                           A41, A42, A43, 0.f, 0.f, elems, hdr);
        hop_gemm_pvf<<<gemm_grid, 256, 0, stream>>>(y, Wsb, b, kb[3], amax, 0, hdr);

        hop_combine<<<CB, CT, 0, stream>>>(y, z, kb[0], kb[1], kb[2], kb[3], kb[0],
                                           A51, A52, A53, A54, 0.f, elems, hdr);
        hop_gemm_pvf<<<gemm_grid, 256, 0, stream>>>(y, Wsb, b, kb[4], amax, 0, hdr);

        hop_combine<<<CB, CT, 0, stream>>>(y, z, kb[0], kb[1], kb[2], kb[3], kb[4],
                                           A61, A62, A63, A64, A65, elems, hdr);
        hop_gemm_pvf<<<gemm_grid, 256, 0, stream>>>(y, Wsb, b, kb[5], amax, 0, hdr);

        // z += DT*(B1*k1 + B3*k3 + B4*k4 + B5*k5 + B6*k6)   (skipped once done)
        hop_combine<<<CB, CT, 0, stream>>>(z, z, kb[0], kb[2], kb[3], kb[4], kb[5],
                                           B1, B3, B4, B5, B6, elems, hdr);
    }
    (void)in_sizes; (void)n_in; (void)out_size; (void)ws_size;
}
